// CondOTGRNModel_51264729645658
// MI455X (gfx1250) — matrix operand order repaired and checked by reading, not compiled, not run on hardware
//
#include <hip/hip_runtime.h>
#include <math.h>

// ---------------------------------------------------------------------------
// CDNA5 / gfx1250: wave32, WMMA bf16 16x16x32, f32 accumulate.
// GEMM operands pre-converted to bf16 (weights pre-transposed to [N,K]);
// tiles move global->LDS via gfx1250 async copies (ASYNCcnt), triple-buffered
// with a 2-deep prefetch pipeline.
// ---------------------------------------------------------------------------

typedef __attribute__((ext_vector_type(16))) __bf16          v16bf;
typedef __attribute__((ext_vector_type(8)))  float           v8f;
typedef __attribute__((ext_vector_type(8)))  unsigned short  su8;

__device__ __forceinline__ unsigned short f2bf(float f) {
  union { float f; unsigned int u; } x; x.f = f;
  unsigned int r = x.u + 0x7FFFu + ((x.u >> 16) & 1u);  // round-to-nearest-even
  return (unsigned short)(r >> 16);
}

__device__ __forceinline__ float gelu_exact(float x) {
  return 0.5f * x * (1.0f + erff(x * 0.70710678118654752440f));
}

// Low 32 bits of a generic shared pointer are the LDS byte address (ISA §10.2).
__device__ __forceinline__ unsigned lds_addr32(const void* p) {
  return (unsigned)(unsigned long long)p;
}

union FragU { v16bf v; su8 h[2]; };

// ---------------------------------------------------------------------------
// Weight prep: W fp32 [K,N] -> Wt bf16 [N,K]   (LDS-tiled transpose)
// ---------------------------------------------------------------------------
__global__ void wt_kernel(const float* __restrict__ W, unsigned short* __restrict__ Wt,
                          int K, int N)
{
  __shared__ float tile[32][33];
  const int n0 = blockIdx.x * 32, k0 = blockIdx.y * 32;
  const int tx = threadIdx.x, ty = threadIdx.y;   // 32 x 8
#pragma unroll
  for (int i = 0; i < 32; i += 8) {
    int k = k0 + ty + i, n = n0 + tx;
    tile[ty + i][tx] = (k < K && n < N) ? W[(size_t)k * N + n] : 0.0f;
  }
  __syncthreads();
#pragma unroll
  for (int i = 0; i < 32; i += 8) {
    int n = n0 + ty + i, k = k0 + tx;
    if (n < N && k < K) Wt[(size_t)n * K + k] = f2bf(tile[tx][ty + i]);
  }
}

// fp32 -> bf16 flat copy
__global__ void f2bf_kernel(const float* __restrict__ x, unsigned short* __restrict__ y, int n)
{
  int i = blockIdx.x * 256 + threadIdx.x;
  if (i < n) y[i] = f2bf(x[i]);
}

// ---------------------------------------------------------------------------
// GEMM: C[M,N] = epi( alpha*(A[M,K] @ B[K,N] + bias) )
//   A: bf16 [M,K] (lda), Bt: bf16 [N,K] pre-transposed (row stride = K).
// REQUIRES: M % 128 == 0, K % 32 == 0, lda % 8 == 0 (all true here).
// Block tile 128x128x32, 256 threads (8 waves), wave = 32x64 via 2x4 WMMAs.
// Triple-buffered LDS, 2-deep async prefetch (GLOBAL_LOAD_ASYNC_TO_LDS_B128,
// ASYNCcnt): each tile = 4 loads/thread, so "s_wait_asynccnt 4" after issuing
// tile t+2 guarantees tile t+1 is resident (in-order completion) while t+2
// stays in flight. Out-of-range B rows are clamped to row N-1 (computed but
// never stored). Epilogue: bias, alpha, exact-GELU, accumulate, residual;
// fp32 or bf16 out.
// ---------------------------------------------------------------------------
__global__ __launch_bounds__(256)
void gemm_kernel(const unsigned short* __restrict__ A, int lda,
                 const unsigned short* __restrict__ Bt,
                 const float* __restrict__ bias,
                 void* __restrict__ Cv, int ldc,
                 int M, int N, int K,
                 int act, float alpha, int accum, int obf,
                 const float* __restrict__ res, int ldr)
{
  // Tiles [128 rows][32 cols] bf16, row stride 40 ushorts (80B) so every 16B
  // fragment chunk stays 16B-aligned and banks are staggered.
  __shared__ unsigned short Ash[3][128 * 40];
  __shared__ unsigned short Bsh[3][128 * 40];

  const int tid  = threadIdx.x;
  const int lane = tid & 31;
  const int wave = tid >> 5;
  const int wm   = wave >> 1;   // 0..3 : M sub-tile (32 rows)
  const int wn   = wave & 1;    // 0..1 : N sub-tile (64 cols)
  const int lh   = lane >> 4;   // lane half (ISA K-split)
  const int lm   = lane & 15;
  const int m0   = blockIdx.y * 128;
  const int n0   = blockIdx.x * 128;

  // Per-thread staging slots: 2 x 16B for A, 2 for B (128*32*2B/16B/256 = 2).
  int vrow[2], vcol[2];
#pragma unroll
  for (int i = 0; i < 2; ++i) {
    int v = tid + 256 * i;          // 0..511 16B slots
    vrow[i] = v >> 2;               // 0..127
    vcol[i] = (v & 3) << 3;         // 0,8,16,24 (elements)
  }
  int brow[2];                      // B rows clamped for the N edge
#pragma unroll
  for (int i = 0; i < 2; ++i) {
    int nr = n0 + vrow[i];
    brow[i] = (nr < N) ? nr : (N - 1);
  }

  // Issue 4 async global->LDS 16B copies for tile at k0 into buffer p.
  auto issue_async = [&](int k0, int p) {
#pragma unroll
    for (int i = 0; i < 2; ++i) {
      unsigned ga = (unsigned)(((size_t)(m0 + vrow[i]) * lda + k0 + vcol[i]) * 2);
      unsigned la = lds_addr32(&Ash[p][vrow[i] * 40 + vcol[i]]);
      asm volatile("global_load_async_to_lds_b128 %0, %1, %2"
                   :: "v"(la), "v"(ga), "s"(A) : "memory");
    }
#pragma unroll
    for (int i = 0; i < 2; ++i) {
      unsigned ga = (unsigned)(((size_t)brow[i] * K + k0 + vcol[i]) * 2);
      unsigned la = lds_addr32(&Bsh[p][vrow[i] * 40 + vcol[i]]);
      asm volatile("global_load_async_to_lds_b128 %0, %1, %2"
                   :: "v"(la), "v"(ga), "s"(Bt) : "memory");
    }
  };

  v8f acc[2][4];
#pragma unroll
  for (int f = 0; f < 2; ++f)
#pragma unroll
    for (int g = 0; g < 4; ++g)
#pragma unroll
      for (int r = 0; r < 8; ++r) acc[f][g][r] = 0.0f;

  const int nk = K >> 5;        // K % 32 == 0
  issue_async(0, 0);
  if (nk > 1) {
    issue_async(32, 1);
    asm volatile("s_wait_asynccnt 4" ::: "memory");   // tile 0 resident
  } else {
    asm volatile("s_wait_asynccnt 0" ::: "memory");
  }
  __syncthreads();

  int p = 0, ib = 2;
  for (int t = 0; t < nk; ++t) {
    if (t + 2 < nk) issue_async((t + 2) << 5, ib);    // uniform branch

    // A 16x32 bf16: lane(half h, m): elem e -> K = 8h+e (e<8) else 8h+e+8
    FragU af[2];
#pragma unroll
    for (int f = 0; f < 2; ++f) {
      const unsigned short* q = &Ash[p][(wm * 32 + f * 16 + lm) * 40 + 8 * lh];
      af[f].h[0] = *(const su8*)(q);
      af[f].h[1] = *(const su8*)(q + 16);
    }
    // B 32x16 bf16: lane(half h, n): elem e -> K = 8h+e (e<8) else 8h+e+8
    FragU bg[4];
#pragma unroll
    for (int g = 0; g < 4; ++g) {
      const unsigned short* q = &Bsh[p][(wn * 64 + g * 16 + lm) * 40 + 8 * lh];
      bg[g].h[0] = *(const su8*)(q);
      bg[g].h[1] = *(const su8*)(q + 16);
    }
#pragma unroll
    for (int f = 0; f < 2; ++f)
#pragma unroll
      for (int g = 0; g < 4; ++g)
        acc[f][g] = __builtin_amdgcn_wmma_f32_16x16x32_bf16(
            false, af[f].v, false, bg[g].v, (short)0, acc[f][g], false, false);

    // Ensure tile t+1 (next to be consumed) has fully landed in LDS.
    if (t + 2 < nk)      asm volatile("s_wait_asynccnt 4" ::: "memory");
    else if (t + 1 < nk) asm volatile("s_wait_asynccnt 0" ::: "memory");
    __syncthreads();
    p  = (p  == 2) ? 0 : p + 1;
    ib = (ib == 2) ? 0 : ib + 1;
  }

  // ---- epilogue: C/D layout -> row = 8*lh + r, col = lm ----
  float* Cf = (float*)Cv;
  unsigned short* Cb = (unsigned short*)Cv;
#pragma unroll
  for (int f = 0; f < 2; ++f) {
#pragma unroll
    for (int g = 0; g < 4; ++g) {
      const int colg = n0 + wn * 64 + g * 16 + lm;
      if (colg >= N) continue;
      const float bv = bias ? bias[colg] : 0.0f;
#pragma unroll
      for (int r = 0; r < 8; ++r) {
        const int rowg = m0 + wm * 32 + f * 16 + 8 * lh + r;  // M % 128 == 0
        float v = (acc[f][g][r] + bv) * alpha;
        if (act == 1) v = gelu_exact(v);
        const size_t ci = (size_t)rowg * ldc + colg;
        if (obf) {
          Cb[ci] = f2bf(v);
        } else {
          if (accum) v += Cf[ci];
          if (res)   v += res[(size_t)rowg * ldr + colg];
          Cf[ci] = v;
        }
      }
    }
  }
}

// ---------------------------------------------------------------------------
// Row LayerNorm, optional fused exact GELU. One block per row.
// wmode bit0: write fp32 in place; bit1: write bf16 to xb.
// ---------------------------------------------------------------------------
__global__ __launch_bounds__(256)
void ln_kernel(float* x, unsigned short* xb, const float* __restrict__ g,
               const float* __restrict__ b, int N, int act, int wmode)
{
  const int row = blockIdx.x;
  float* xr = x + (size_t)row * N;
  unsigned short* xbr = xb ? xb + (size_t)row * N : nullptr;
  float s = 0.f, q = 0.f;
  for (int j = threadIdx.x; j < N; j += 256) { float v = xr[j]; s += v; q += v * v; }
  __shared__ float sb[256], qb[256];
  sb[threadIdx.x] = s; qb[threadIdx.x] = q;
  __syncthreads();
  for (int o = 128; o > 0; o >>= 1) {
    if (threadIdx.x < o) { sb[threadIdx.x] += sb[threadIdx.x + o]; qb[threadIdx.x] += qb[threadIdx.x + o]; }
    __syncthreads();
  }
  float mean = sb[0] / (float)N;
  float var  = qb[0] / (float)N - mean * mean;
  float rstd = rsqrtf(var + 1e-5f);
  for (int j = threadIdx.x; j < N; j += 256) {
    float v = (xr[j] - mean) * rstd * g[j] + b[j];
    if (act) v = gelu_exact(v);
    if (wmode & 1) xr[j]  = v;
    if (wmode & 2) xbr[j] = f2bf(v);
  }
}

// dose -> gelu(d*W1+b1) -> gelu(@W2+b2) -> bf16 ; one wave per row.
__global__ void dose_kernel(const float* __restrict__ dose,
                            const float* __restrict__ W1, const float* __restrict__ b1,
                            const float* __restrict__ W2, const float* __restrict__ b2,
                            unsigned short* __restrict__ out)
{
  const int row = blockIdx.x, j = threadIdx.x;
  __shared__ float h[32];
  float d = dose[row];
  h[j] = gelu_exact(d * W1[j] + b1[j]);
  __syncthreads();
  float a = b2[j];
  for (int i = 0; i < 32; ++i) a += h[i] * W2[i * 32 + j];
  out[row * 32 + j] = f2bf(gelu_exact(a));
}

__global__ void concat_kernel(const unsigned short* __restrict__ a, int na,
                              const unsigned short* __restrict__ b, int nb,
                              unsigned short* __restrict__ o)
{
  const int row = blockIdx.x, n = na + nb;
  for (int j = threadIdx.x; j < n; j += blockDim.x)
    o[(size_t)row * n + j] = (j < na) ? a[(size_t)row * na + j]
                                      : b[(size_t)row * nb + (j - na)];
}

// zr = z + noise * 0.1*||p_row||  (block = 128 threads per row)
__global__ void latent_kernel(const float* __restrict__ z, const float* __restrict__ noise,
                              const float* __restrict__ p, float* __restrict__ zr)
{
  const int row = blockIdx.x, tid = threadIdx.x;
  __shared__ float sb[64];
  if (tid < 64) { float v = p[row * 64 + tid]; sb[tid] = v * v; }
  __syncthreads();
  for (int o = 32; o > 0; o >>= 1) { if (tid < o) sb[tid] += sb[tid + o]; __syncthreads(); }
  float ns = 0.1f * sqrtf(sb[0]);
  zr[row * 128 + tid] = z[row * 128 + tid] + noise[row * 128 + tid] * ns;
}

// si[:, :64] = z1 = zr*(1-mask) first half ; si[:, 64:] = p   (bf16 out)
__global__ void flow_pre_kernel(const float* __restrict__ zr, const float* __restrict__ p,
                                unsigned short* __restrict__ si, int par)
{
  const int row = blockIdx.x, j = threadIdx.x; // 128 threads
  float v;
  if (j < 64) v = ((j & 1) == par) ? 0.0f : zr[row * 128 + j];
  else        v = p[row * 64 + (j - 64)];
  si[row * 128 + j] = f2bf(v);
}

// second-half unmasked entries: zr = zr*exp(tanh(s)) + t   (exact mask algebra)
__global__ void flow_post_kernel(float* __restrict__ zr, const float* __restrict__ slin,
                                 const float* __restrict__ tlin, int par)
{
  const int row = blockIdx.x, j = threadIdx.x; // 64 threads
  const int j2 = 64 + j;
  if ((j2 & 1) != par) {
    float s = tanhf(slin[row * 64 + j]);
    zr[row * 128 + j2] = zr[row * 128 + j2] * expf(s) + tlin[row * 64 + j];
  }
}

// ---------------------------------------------------------------------------
// Host orchestration.
// Param leaf order assumes jax tree_flatten (dict keys sorted; 'W' < 'b';
// _N -> b then g) after the 4 array inputs:
//   chem[0..15], dec[16..85], flow[86..133], ge[134..149], head[150..155],
//   inproj[156..167]
// ---------------------------------------------------------------------------
extern "C" void kernel_launch(void* const* d_in, const int* in_sizes, int n_in,
                              void* d_out, int out_size, void* d_ws, size_t ws_size,
                              hipStream_t stream)
{
  (void)in_sizes; (void)n_in; (void)out_size; (void)ws_size;
  const float* x_control = (const float*)d_in[0];   // [2048,20000]
  const float* pert      = (const float*)d_in[1];   // [2048,2048]
  const float* dose      = (const float*)d_in[2];   // [2048,1]
  const float* noise     = (const float*)d_in[3];   // [2048,128]
  const float* const* P  = (const float* const*)(d_in + 4);

  // ---- workspace bump allocator ----
  char* base = (char*)d_ws; size_t woff = 0;
  auto alloc = [&](size_t bytes) -> void* {
    void* r = (void*)(base + woff);
    woff += (bytes + 255) & ~(size_t)255;
    return r;
  };

  // fp32 temps
  float* t2048 = (float*)alloc(2048ull * 2048 * 4);
  float* t512a = (float*)alloc(2048ull * 512 * 4);
  float* t512b = (float*)alloc(2048ull * 512 * 4);
  float* t512c = (float*)alloc(2048ull * 512 * 4);
  float* t256  = (float*)alloc(2048ull * 256 * 4);
  float* t128a = (float*)alloc(2048ull * 128 * 4);   // x_pred_proj
  float* t128b = (float*)alloc(2048ull * 128 * 4);   // z
  float* t128c = (float*)alloc(2048ull * 128 * 4);   // zr
  float* t64p  = (float*)alloc(2048ull * 64 * 4);    // p
  float* t64s  = (float*)alloc(2048ull * 64 * 4);
  float* t64t  = (float*)alloc(2048ull * 64 * 4);

  // bf16 activation buffers
  unsigned short* bx    = (unsigned short*)alloc(2048ull * 20000 * 2);
  unsigned short* bpert = (unsigned short*)alloc(2048ull * 2048 * 2);
  unsigned short* b2048 = (unsigned short*)alloc(2048ull * 2048 * 2);
  unsigned short* b512  = (unsigned short*)alloc(2048ull * 512 * 2);
  unsigned short* b256  = (unsigned short*)alloc(2048ull * 256 * 2);
  unsigned short* b288  = (unsigned short*)alloc(2048ull * 288 * 2);
  unsigned short* b128x = (unsigned short*)alloc(2048ull * 128 * 2);  // h / zr / xpp
  unsigned short* b128s = (unsigned short*)alloc(2048ull * 128 * 2);  // si
  unsigned short* b32d  = (unsigned short*)alloc(2048ull * 32 * 2);   // df

  // ---- pre-pass: weights fp32 [K,N] -> bf16 [N,K] ----
  auto prep_w = [&](int idx, int K, int N) -> const unsigned short* {
    unsigned short* dst = (unsigned short*)alloc((size_t)K * N * 2);
    dim3 g((N + 31) / 32, (K + 31) / 32);
    wt_kernel<<<g, dim3(32, 8), 0, stream>>>(P[idx], dst, K, N);
    return dst;
  };
  const unsigned short* Wl1  = prep_w(156, 20000, 2048);
  const unsigned short* Wl2  = prep_w(158, 2048, 512);
  const unsigned short* Wl3  = prep_w(160, 512, 128);
  const unsigned short* Winp = prep_w(134, 128, 512);
  const unsigned short* Wm0  = prep_w(136, 512, 512);
  const unsigned short* Wm1g = prep_w(140, 512, 512);
  const unsigned short* Wo1  = prep_w(144, 512, 256);
  const unsigned short* Wo2  = prep_w(148, 256, 128);
  const unsigned short* Wp1  = prep_w(8,   2048, 512);
  const unsigned short* Wp2  = prep_w(12,  512, 256);
  const unsigned short* Wfu  = prep_w(4,   288, 64);
  const unsigned short* Wfl[6][4];
  for (int i = 0; i < 6; ++i) {
    Wfl[i][0] = prep_w(86 + 8 * i + 0, 128, 512);
    Wfl[i][1] = prep_w(86 + 8 * i + 2, 512, 64);
    Wfl[i][2] = prep_w(86 + 8 * i + 4, 128, 512);
    Wfl[i][3] = prep_w(86 + 8 * i + 6, 512, 64);
  }
  const unsigned short* Wdm1  = prep_w(78, 128, 512);
  const unsigned short* Wdm2  = prep_w(82, 512, 512);
  const unsigned short* Wgres = prep_w(16, 128, 128);
  const unsigned short* Whh1[10];
  const unsigned short* Whh2[10];
  for (int i = 0; i < 10; ++i) {
    Whh1[i] = prep_w(18 + 6 * i + 0, 512, 256);
    Whh2[i] = prep_w(18 + 6 * i + 4, 256, (i < 9) ? 12 : 20);
  }
  const unsigned short* Woh1 = prep_w(150, 128, 512);
  const unsigned short* Woh2 = prep_w(154, 512, 20000);

  // activations -> bf16
  f2bf_kernel<<<(2048 * 20000 + 255) / 256, 256, 0, stream>>>(x_control, bx, 2048 * 20000);
  f2bf_kernel<<<(2048 * 2048 + 255) / 256, 256, 0, stream>>>(pert, bpert, 2048 * 2048);

  auto gemm = [&](const unsigned short* A, int lda, const unsigned short* Bt,
                  const float* bias, void* C, int ldc, int N, int K,
                  int act, float alpha, int accum, int obf,
                  const float* r, int ldr) {
    dim3 grid((N + 127) / 128, 16);   // M = 2048
    gemm_kernel<<<grid, 256, 0, stream>>>(A, lda, Bt, bias, C, ldc, 2048, N, K,
                                          act, alpha, accum, obf, r, ldr);
  };
  auto ln = [&](float* x, unsigned short* xb, const float* g, const float* b,
                int N, int act, int wmode) {
    ln_kernel<<<2048, 256, 0, stream>>>(x, xb, g, b, N, act, wmode);
  };

  // ---- inproj ----
  gemm(bx, 20000, Wl1, P[157], t2048, 2048, 2048, 20000, 0, 1.f, 0, 0, nullptr, 0);
  ln(t2048, b2048, P[163], P[162], 2048, 1, 2);
  gemm(b2048, 2048, Wl2, P[159], t512a, 512, 512, 2048, 0, 1.f, 0, 0, nullptr, 0);
  ln(t512a, b512, P[165], P[164], 512, 1, 2);
  gemm(b512, 512, Wl3, P[161], t128a, 128, 128, 512, 0, 1.f, 0, 0, nullptr, 0);
  ln(t128a, b128x, P[167], P[166], 128, 0, 2);                       // h (bf16)

  // ---- gene encoder ----
  gemm(b128x, 128, Winp, P[135], b512, 512, 512, 128, 0, 1.f, 0, 1, nullptr, 0); // g (bf16 out)
  gemm(b512, 512, Wm0, P[137], t512c, 512, 512, 512, 0, 1.f, 0, 0, nullptr, 0);
  ln(t512c, b512, P[139], P[138], 512, 1, 2);
  gemm(b512, 512, Wm1g, P[141], t512c, 512, 512, 512, 0, 1.f, 0, 0, nullptr, 0);
  ln(t512c, b512, P[143], P[142], 512, 1, 2);
  gemm(b512, 512, Wo1, P[145], t256, 256, 256, 512, 0, 1.f, 0, 0, nullptr, 0);
  ln(t256, b256, P[147], P[146], 256, 1, 2);
  gemm(b256, 256, Wo2, P[149], t128b, 128, 128, 256, 0, 1.f, 0, 0, nullptr, 0);  // z (fp32)

  // ---- chem encoder ----
  gemm(bpert, 2048, Wp1, P[9], t512c, 512, 512, 2048, 0, 1.f, 0, 0, nullptr, 0);
  ln(t512c, b512, P[11], P[10], 512, 1, 2);
  gemm(b512, 512, Wp2, P[13], t256, 256, 256, 512, 0, 1.f, 0, 0, nullptr, 0);
  ln(t256, b256, P[15], P[14], 256, 1, 2);
  dose_kernel<<<2048, 32, 0, stream>>>(dose, P[0], P[1], P[2], P[3], b32d);
  concat_kernel<<<2048, 288, 0, stream>>>(b256, 256, b32d, 32, b288);
  gemm(b288, 288, Wfu, P[5], t64p, 64, 64, 288, 0, 1.f, 0, 0, nullptr, 0);
  ln(t64p, nullptr, P[7], P[6], 64, 1, 1);                           // p (fp32 only)

  // ---- latent perturbation ----
  latent_kernel<<<2048, 128, 0, stream>>>(t128b, noise, t64p, t128c);  // zr

  // ---- flow: 6 affine coupling layers ----
  for (int i = 0; i < 6; ++i) {
    const int par = i & 1;
    const int fb  = 86 + 8 * i;
    flow_pre_kernel<<<2048, 128, 0, stream>>>(t128c, t64p, b128s, par);
    gemm(b128s, 128, Wfl[i][0], P[fb + 1], b512, 512, 512, 128, 1, 1.f, 0, 1, nullptr, 0);
    gemm(b512, 512, Wfl[i][1], P[fb + 3], t64s, 64, 64, 512, 0, 1.f, 0, 0, nullptr, 0);
    gemm(b128s, 128, Wfl[i][2], P[fb + 5], b512, 512, 512, 128, 1, 1.f, 0, 1, nullptr, 0);
    gemm(b512, 512, Wfl[i][3], P[fb + 7], t64t, 64, 64, 512, 0, 1.f, 0, 0, nullptr, 0);
    flow_post_kernel<<<2048, 64, 0, stream>>>(t128c, t64s, t64t, par);
  }

  // ---- gene decoder ----
  f2bf_kernel<<<(2048 * 128 + 255) / 256, 256, 0, stream>>>(t128c, b128x, 2048 * 128); // zr bf16
  gemm(b128x, 128, Wdm1, P[79], t512b, 512, 512, 128, 0, 1.f, 0, 0, nullptr, 0);
  ln(t512b, b512, P[81], P[80], 512, 1, 2);
  gemm(b512, 512, Wdm2, P[83], t512c, 512, 512, 512, 0, 1.f, 0, 0, nullptr, 0);
  ln(t512c, b512, P[85], P[84], 512, 1, 2);                          // mf (bf16 in b512)
  gemm(b128x, 128, Wgres, P[17], t128a, 128, 128, 128, 0, 0.1f, 0, 0, nullptr, 0);
  {
    int offc = 0;
    for (int i = 0; i < 10; ++i) {
      const int hb = 18 + 6 * i;
      const int sz = (i < 9) ? 12 : 20;
      gemm(b512, 512, Whh1[i], P[hb + 1], t256, 256, 256, 512, 0, 1.f, 0, 0, nullptr, 0);
      ln(t256, b256, P[hb + 3], P[hb + 2], 256, 1, 2);
      gemm(b256, 256, Whh2[i], P[hb + 5], t128a + offc, 128, sz, 256, 0, 1.f, 1, 0, nullptr, 0);
      offc += sz;
    }
  }

  // ---- output head (residual add of x_control fused) ----
  f2bf_kernel<<<(2048 * 128 + 255) / 256, 256, 0, stream>>>(t128a, b128x, 2048 * 128); // xpp bf16
  gemm(b128x, 128, Woh1, P[151], t512b, 512, 512, 128, 0, 1.f, 0, 0, nullptr, 0);
  ln(t512b, b512, P[153], P[152], 512, 1, 2);
  gemm(b512, 512, Woh2, P[155], d_out, 20000, 20000, 512, 0, 1.f, 0, 0, x_control, 20000);
}
